// Siamese_lstm_38946763440222
// MI455X (gfx1250) — matrix operand order repaired and checked by reading, not compiled, not run on hardware
//
#include <hip/hip_runtime.h>

// ---------------------------------------------------------------------------
// Siamese LSTM for MI455X (gfx1250, wave32, WMMA + TDM).
//   V=50000 E=1024 H=1024 L=2048 FC=512, 4H=4096, 2 sequences.
// Pipeline:
//   k_prep_weights : W_ih,W_hh -> bf16 ; bias = b_ih+b_hh (f32)
//   k_embed        : xs[2L,E] = bf16(emb[tokens])
//   k_gemm_gx      : gx[2L,4H] = xs @ W_ih^T + bias   (WMMA bf16, f32 acc)
//   k_lstm         : persistent, 64 WGs; W_hh slice (128KB bf16) loaded into
//                    LDS via TENSOR_LOAD_TO_LDS (TDM); per step one 64x1024
//                    matvec via WMMA with both sequences as 2 B-columns;
//                    grid barrier via atomics.
//   k_head         : feat -> fc1 -> fc2 -> d_out[2] (f32)
// ---------------------------------------------------------------------------

typedef __bf16 bf16_t;
typedef bf16_t v16bf __attribute__((ext_vector_type(16)));
typedef bf16_t v8bf  __attribute__((ext_vector_type(8)));
typedef float  v8f   __attribute__((ext_vector_type(8)));
typedef unsigned int u32x4 __attribute__((ext_vector_type(4)));
typedef int          i32x8 __attribute__((ext_vector_type(8)));
typedef int          i32x4 __attribute__((ext_vector_type(4)));

#define E_    1024
#define H_    1024
#define L_    2048
#define FC_   512
#define G4_   4096            // 4*H
#define MTOT_ 4096            // 2*L
#define NWG_  64              // workgroups in persistent LSTM kernel

// workspace layout (bytes)
static constexpr size_t XS_OFF   = 0;                                   // 8 MB bf16
static constexpr size_t WIH_OFF  = XS_OFF  + (size_t)MTOT_ * E_ * 2;    // 8 MB bf16
static constexpr size_t WHH_OFF  = WIH_OFF + (size_t)G4_  * E_ * 2;     // 8 MB bf16
static constexpr size_t BIAS_OFF = WHH_OFF + (size_t)G4_  * H_ * 2;     // 16 KB f32
static constexpr size_t HBUF_OFF = BIAS_OFF + (size_t)G4_ * 4;          // 2par*2seq*H bf16
static constexpr size_t VFIN_OFF = HBUF_OFF + 2ull * 2 * H_ * 2;        // 2seq*H f32
static constexpr size_t CNT_OFF  = (VFIN_OFF + 2ull * H_ * 4 + 255) & ~255ull;
static constexpr size_t GX_OFF   = (CNT_OFF + 256 + 255) & ~255ull;     // 64 MB f32

__device__ __forceinline__ float sigm(float x) { return 1.0f / (1.0f + expf(-x)); }

__device__ __forceinline__ v16bf pack16(v8bf lo, v8bf hi) {
  v16bf r;
#pragma unroll
  for (int i = 0; i < 8; ++i) { r[i] = lo[i]; r[i + 8] = hi[i]; }
  return r;
}

// ---------------------------------------------------------------------------
__global__ void __launch_bounds__(256)
k_prep_weights(const float* __restrict__ wih, const float* __restrict__ whh,
               const float* __restrict__ bih, const float* __restrict__ bhh,
               bf16_t* __restrict__ wih_b, bf16_t* __restrict__ whh_b,
               float* __restrict__ bias) {
  size_t i = (size_t)blockIdx.x * 256 + threadIdx.x;
  const size_t n = (size_t)G4_ * E_;                 // 4,194,304
  if (i < n)            wih_b[i]     = (bf16_t)wih[i];
  else if (i < 2 * n)   whh_b[i - n] = (bf16_t)whh[i - n];
  if (i < (size_t)G4_)  bias[i] = bih[i] + bhh[i];
}

// ---------------------------------------------------------------------------
__global__ void __launch_bounds__(256)
k_embed(const int* __restrict__ s1, const int* __restrict__ s2,
        const float* __restrict__ emb, bf16_t* __restrict__ xs) {
  size_t i = (size_t)blockIdx.x * 256 + threadIdx.x;  // over MTOT_*E_
  int row = (int)(i >> 10);                           // [0,4096)
  int e   = (int)(i & 1023);
  int seq = row >> 11;                                // row = seq*L + t
  int t   = row & 2047;
  int tok = seq ? s2[t] : s1[t];
  xs[i] = (bf16_t)emb[(size_t)tok * E_ + e];
}

// ---------------------------------------------------------------------------
// gx[m,n] = sum_k xs[m,k] * W_ih[n,k] + bias[n]
// One 16x16 output tile per wave, K loop of 32 wmma_f32_16x16x32_bf16.
__global__ void __launch_bounds__(256)
k_gemm_gx(const bf16_t* __restrict__ xs, const bf16_t* __restrict__ wih,
          const float* __restrict__ bias, float* __restrict__ gx) {
  int lane = threadIdx.x & 31;
  int wave = threadIdx.x >> 5;
  int tile = blockIdx.x * 8 + wave;     // 65536 tiles = 256(m) x 256(n)
  int nt = tile & 255, mt = tile >> 8;
  int m0 = mt * 16, n0 = nt * 16;

  int mA = m0 + (lane & 15);
  int aK = (lane >> 4) * 8;             // A K-half select per ISA layout
  int nB = n0 + (lane & 15);
  int bK = (lane >> 4) * 8;             // B K-half select per ISA layout

  const bf16_t* arow = xs  + (size_t)mA * E_;
  const bf16_t* brow = wih + (size_t)nB * E_;

  v8f acc = {};
#pragma unroll 4
  for (int k = 0; k < E_; k += 32) {
    __builtin_prefetch(arow + k + 256, 0, 1);   // global_prefetch_b8
    v8bf a0 = *(const v8bf*)(arow + k + aK);
    v8bf a1 = *(const v8bf*)(arow + k + 16 + aK);
    v16bf a = pack16(a0, a1);
    v16bf b = pack16(*(const v8bf*)(brow + k + bK), *(const v8bf*)(brow + k + 16 + bK));
    acc = __builtin_amdgcn_wmma_f32_16x16x32_bf16(false, a, false, b,
                                                  (short)0, acc, false, false);
  }
  // D layout: VGPR r, lane l -> M = r + 8*(l>>4), N = l&15
  int mrow = (lane >> 4) * 8;
  int n    = n0 + (lane & 15);
  float bn = bias[n];
#pragma unroll
  for (int r = 0; r < 8; ++r)
    gx[(size_t)(m0 + mrow + r) * G4_ + n] = acc[r] + bn;
}

// ---------------------------------------------------------------------------
// Persistent LSTM. WG wg owns hidden slice j in [wg*16, wg*16+16).
// LDS holds its 64 W_hh rows (4 gates x 16 rows x 1024 K, bf16 = 128 KB),
// loaded by the Tensor Data Mover (one 2D-tile descriptor per gate).
// 8 waves: wave = gate*2 + khalf. Both sequences ride as B columns 0 and 1.
__global__ void __launch_bounds__(256)
k_lstm(const bf16_t* __restrict__ whh, const float* __restrict__ gx,
       const float* __restrict__ h01, const float* __restrict__ c01,
       const float* __restrict__ h02, const float* __restrict__ c02,
       bf16_t* __restrict__ hbuf, float* __restrict__ vfin,
       unsigned* __restrict__ cnt) {
  extern __shared__ char smem[];
  bf16_t* wlds = (bf16_t*)smem;                       // [64][1024] bf16
  float*  part = (float*)(smem + 64 * 1024 * 2);      // [8 waves][16 m][2 seq]
  float*  csta = part + 8 * 16 * 2;                   // [2 seq][16]

  const int tid  = threadIdx.x;
  const int lane = tid & 31;
  const int wave = tid >> 5;
  const int wg   = blockIdx.x;

  // ---- preload W_hh slice into LDS ----
#if defined(__gfx1250__) && __has_builtin(__builtin_amdgcn_tensor_load_to_lds)
  // TDM path: one 2D tile per gate: 16 rows x 2048B, row stride 2048B.
  // D# per CDNA5 ISA; data_size=3 (8B units):
  //   tensor_dim0 = 256 (8B words/row), tensor_dim1 = 4096 rows,
  //   tile_dim0 = 256, tile_dim1 = 16, tensor_dim0_stride = 256.
  if (wave == 0 && lane == 0) {
    unsigned long long wbase = (unsigned long long)(uintptr_t)whh;
#pragma unroll
    for (int gg = 0; gg < 4; ++gg) {
      unsigned long long gaddr =
          wbase + (unsigned long long)(gg * H_ + wg * 16) * (H_ * 2);
      u32x4 d0;
      d0[0] = 1u;                                   // count=1, user descriptor
      d0[1] = (unsigned)(gg * 16 * H_ * 2);         // lds_addr (bytes)
      d0[2] = (unsigned)(gaddr & 0xffffffffu);      // global_addr[31:0]
      d0[3] = (unsigned)((gaddr >> 32) & 0x1ffffffu) | (2u << 30); // [56:32]|type=2
      i32x8 d1;
      d1[0] = (3 << 16);                // wg_mask=0, data_size=3 (8B)
      d1[1] = (int)(256u << 16);        // tensor_dim0[15:0] = 256
      d1[2] = (int)(4096u << 16);       // tensor_dim1[15:0] = 4096
      d1[3] = (int)(256u << 16);        // tile_dim0 = 256
      d1[4] = 16;                       // tile_dim1 = 16, tile_dim2 = 0
      d1[5] = 256;                      // tensor_dim0_stride[31:0] = 256
      d1[6] = 0;                        // dim0_stride[47:32], dim1_stride lo
      d1[7] = 0;
      i32x4 dz4 = {0, 0, 0, 0};
      i32x8 dz8 = {0, 0, 0, 0, 0, 0, 0, 0};
      __builtin_amdgcn_tensor_load_to_lds(d0, d1, dz4, dz4, dz8, 0);
    }
    __builtin_amdgcn_s_wait_tensorcnt(0);           // s_wait_tensorcnt 0
  }
#else
  // Fallback: cooperative VGPR copy (uint32 words).
  for (int i = tid; i < 64 * (H_ / 2); i += 256) {
    int row  = i / (H_ / 2);
    int word = i % (H_ / 2);
    int g = row >> 4, m = row & 15;
    size_t R = (size_t)(g * H_ + wg * 16 + m);
    ((uint32_t*)wlds)[i] = ((const uint32_t*)whh)[R * (H_ / 2) + word];
  }
#endif
  // ---- init c (LDS, f32) and h parity-0 buffer (global, bf16) ----
  if (tid < 32) {
    int seq = tid >> 4, jj = tid & 15, j = wg * 16 + jj;
    csta[seq * 16 + jj] = seq ? c02[j] : c01[j];
    float h0 = seq ? h02[j] : h01[j];
    hbuf[(size_t)(0 * 2 + seq) * H_ + j] = (bf16_t)h0;
  }
  __threadfence();
  __syncthreads();
  if (tid == 0)
    __hip_atomic_fetch_add(cnt, 1u, __ATOMIC_ACQ_REL, __HIP_MEMORY_SCOPE_AGENT);
  while (__hip_atomic_load(cnt, __ATOMIC_ACQUIRE, __HIP_MEMORY_SCOPE_AGENT) <
         (unsigned)NWG_)
    __builtin_amdgcn_s_sleep(1);
  __syncthreads();

  const int g     = wave >> 1;
  const int khalf = wave & 1;
  const int mA = g * 16 + (lane & 15);     // row within 64-row slice
  const int aK = (lane >> 4) * 8;
  const int nB = lane & 15;                // B column (0 = seq1, 1 = seq2)
  const int bK = (lane >> 4) * 8;

  for (int t = 0; t < L_; ++t) {
    const int p = t & 1;
    const bf16_t* hread = hbuf + (size_t)p * 2 * H_;
    const bf16_t* arow  = wlds + (size_t)mA * H_ + khalf * 512;
    const bf16_t* bbase = (nB < 2) ? (hread + (size_t)nB * H_ + khalf * 512)
                                   : (const bf16_t*)nullptr;
    v8f acc = {};
#pragma unroll 4
    for (int k = 0; k < 512; k += 32) {
      v8bf a0 = *(const v8bf*)(arow + k + aK);       // ds_load from LDS
      v8bf a1 = *(const v8bf*)(arow + k + 16 + aK);
      v16bf a = pack16(a0, a1);
      v16bf b = {};
      if (nB < 2) b = pack16(*(const v8bf*)(bbase + k + bK), *(const v8bf*)(bbase + k + 16 + bK));
      acc = __builtin_amdgcn_wmma_f32_16x16x32_bf16(false, a, false, b,
                                                    (short)0, acc, false, false);
    }
    // stash columns 0/1 of this wave's D tile
    if ((lane & 15) < 2) {
      int mrow = (lane >> 4) * 8;
#pragma unroll
      for (int r = 0; r < 8; ++r)
        part[(wave * 16 + mrow + r) * 2 + nB] = acc[r];
    }
    __syncthreads();

    // pointwise gate update: 16 j-slots x 2 sequences on 32 lanes
    if (tid < 32) {
      int seq = tid >> 4, jj = tid & 15, j = wg * 16 + jj;
      size_t gbase = (size_t)(seq * L_ + t) * G4_;
      float gv[4];
#pragma unroll
      for (int gg = 0; gg < 4; ++gg)
        gv[gg] = gx[gbase + gg * H_ + j] +
                 part[((gg * 2 + 0) * 16 + jj) * 2 + seq] +
                 part[((gg * 2 + 1) * 16 + jj) * 2 + seq];
      float i_ = sigm(gv[0]);
      float f_ = sigm(gv[1]);
      float z_ = tanhf(gv[2]);
      float o_ = sigm(gv[3]);
      float c  = f_ * csta[seq * 16 + jj] + i_ * z_;
      csta[seq * 16 + jj] = c;
      float h = o_ * tanhf(c);
      hbuf[(size_t)((1 - p) * 2 + seq) * H_ + j] = (bf16_t)h;
      if (t == L_ - 1) vfin[(size_t)seq * H_ + j] = h;
    }
    __threadfence();
    __syncthreads();
    if (tid == 0)
      __hip_atomic_fetch_add(cnt, 1u, __ATOMIC_ACQ_REL, __HIP_MEMORY_SCOPE_AGENT);
    unsigned target = (unsigned)NWG_ * (unsigned)(t + 2);
    while (__hip_atomic_load(cnt, __ATOMIC_ACQUIRE, __HIP_MEMORY_SCOPE_AGENT) <
           target)
      __builtin_amdgcn_s_sleep(1);
    __syncthreads();
  }
}

// ---------------------------------------------------------------------------
__global__ void __launch_bounds__(256)
k_head(const float* __restrict__ vfin, const float* __restrict__ fc1w,
       const float* __restrict__ fc1b, const float* __restrict__ fc2w,
       const float* __restrict__ fc2b, float* __restrict__ out) {
  __shared__ float fc1o[FC_];
  const float* v1 = vfin;
  const float* v2 = vfin + H_;
  int tid = threadIdx.x;
  for (int o = tid; o < FC_; o += 256) {
    const float* wrow = fc1w + (size_t)o * G4_;
    float s = fc1b[o];
    for (int k = 0; k < H_; ++k) {
      float a = v1[k], b = v2[k];
      s += wrow[k] * a + wrow[H_ + k] * b +
           wrow[2 * H_ + k] * fabsf(a - b) + wrow[3 * H_ + k] * (a * b);
    }
    fc1o[o] = s;
  }
  __syncthreads();
  if (tid < 2) {
    const float* wrow = fc2w + tid * FC_;
    float s = fc2b[tid];
    for (int k = 0; k < FC_; ++k) s += wrow[k] * fc1o[k];
    out[tid] = s;
  }
}

// ---------------------------------------------------------------------------
extern "C" void kernel_launch(void* const* d_in, const int* in_sizes, int n_in,
                              void* d_out, int out_size, void* d_ws,
                              size_t ws_size, hipStream_t stream) {
  (void)in_sizes; (void)n_in; (void)out_size; (void)ws_size;
  const int*   s1   = (const int*)d_in[0];
  const int*   s2   = (const int*)d_in[1];
  const float* emb  = (const float*)d_in[2];
  const float* wih  = (const float*)d_in[3];
  const float* whh  = (const float*)d_in[4];
  const float* bih  = (const float*)d_in[5];
  const float* bhh  = (const float*)d_in[6];
  const float* h01  = (const float*)d_in[7];
  const float* c01  = (const float*)d_in[8];
  const float* h02  = (const float*)d_in[9];
  const float* c02  = (const float*)d_in[10];
  const float* fc1w = (const float*)d_in[11];
  const float* fc1b = (const float*)d_in[12];
  const float* fc2w = (const float*)d_in[13];
  const float* fc2b = (const float*)d_in[14];

  char* ws = (char*)d_ws;
  bf16_t*   xs    = (bf16_t*)(ws + XS_OFF);
  bf16_t*   wih_b = (bf16_t*)(ws + WIH_OFF);
  bf16_t*   whh_b = (bf16_t*)(ws + WHH_OFF);
  float*    bias  = (float*)(ws + BIAS_OFF);
  bf16_t*   hbuf  = (bf16_t*)(ws + HBUF_OFF);
  float*    vfin  = (float*)(ws + VFIN_OFF);
  unsigned* cnt   = (unsigned*)(ws + CNT_OFF);
  float*    gx    = (float*)(ws + GX_OFF);

  (void)hipMemsetAsync(ws + CNT_OFF, 0, 256, stream);

  k_prep_weights<<<32768, 256, 0, stream>>>(wih, whh, bih, bhh, wih_b, whh_b, bias);
  k_embed<<<16384, 256, 0, stream>>>(s1, s2, emb, xs);
  k_gemm_gx<<<8192, 256, 0, stream>>>(xs, wih_b, bias, gx);

  size_t lstm_smem = 64 * 1024 * 2 + (8 * 16 * 2 + 2 * 16) * sizeof(float);
  k_lstm<<<NWG_, 256, lstm_smem, stream>>>(whh_b, gx, h01, c01, h02, c02,
                                           hbuf, vfin, cnt);
  k_head<<<1, 256, 0, stream>>>(vfin, fc1w, fc1b, fc2w, fc2b, (float*)d_out);
}
